// LSHAtt_37383395344514
// MI455X (gfx1250) — matrix operand order repaired and checked by reading, not compiled, not run on hardware
//
#include <hip/hip_runtime.h>
#include <hip/hip_bf16.h>
#include <cstdint>
#include <cstddef>

// Problem constants (match reference)
#define NPTS 65536
#define HEADS 8
#define HD 64          // per-head dim
#define KD 5           // RPE coord dims
#define BLK 128        // LSH block size
#define HDIM 512       // HEADS*HD
#define SORT_CHUNK 2048

typedef __attribute__((ext_vector_type(16))) _Float16 v16h;
typedef __attribute__((ext_vector_type(8)))  _Float16 v8h;
typedef __attribute__((ext_vector_type(8)))  float    v8f;

// ---------------------------------------------------------------------------
// WMMA fragment loaders (wave32, 16x16x32 f16 -> f32)
// A (16x32, MxK): lane l -> row M=l&15; K chunk base kb=(l>>4)*8;
//   v[0..7] hold K=[kb,kb+8), v[8..15] hold K=[kb+16,kb+24).
// B (32x16, KxN): lane l -> col N=l&15; K chunk base kb=(l>>4)*8; same K split as A.
// C/D (16x16 f32): VGPR r, lane l -> M = r + 8*(l>>4), N = l&15.
// ---------------------------------------------------------------------------
__device__ __forceinline__ v16h frag_a(const _Float16* p0, int ld) {
  int l = threadIdx.x & 31;
  const _Float16* p = p0 + (l & 15) * ld + ((l >> 4) << 3);
  v8h lo = *(const v8h*)(p);
  v8h hi = *(const v8h*)(p + 16);
  v16h a;
#pragma unroll
  for (int i = 0; i < 8; ++i) { a[i] = lo[i]; a[i + 8] = hi[i]; }
  return a;
}

__device__ __forceinline__ v16h frag_b(const _Float16* p0, int ld) {
  int l = threadIdx.x & 31;
  const _Float16* p = p0 + (l & 15) * ld + ((l >> 4) << 3);
  v8h lo = *(const v8h*)(p);
  v8h hi = *(const v8h*)(p + 16);
  v16h b;
#pragma unroll
  for (int i = 0; i < 8; ++i) { b[i] = lo[i]; b[i + 8] = hi[i]; }
  return b;
}

#define WMMA_F16(a, b, c) \
  __builtin_amdgcn_wmma_f32_16x16x32_f16(false, (a), false, (b), (short)0, (c), false, false)

// ---------------------------------------------------------------------------
// Weight conversion (6 matrices) + w2 RPE reduction, one launch.
// Last block computes w2; others convert f32 -> f16.
// ---------------------------------------------------------------------------
__global__ void prep_kernel(
    const float* __restrict__ wq, const float* __restrict__ wk,
    const float* __restrict__ wv, const float* __restrict__ wo,
    const float* __restrict__ f1, const float* __restrict__ f2,
    const float* __restrict__ w_rpe,
    _Float16* __restrict__ owq, _Float16* __restrict__ owk,
    _Float16* __restrict__ owv, _Float16* __restrict__ owo,
    _Float16* __restrict__ of1, _Float16* __restrict__ of2,
    float* __restrict__ w2) {
  if (blockIdx.x == gridDim.x - 1) {
    int idx = threadIdx.x;  // w2[h][k] = mean_d sum_r w_rpe[(h*64+d), r*K+k]^2
    if (idx < HEADS * KD) {
      int h = idx / KD, kk = idx % KD;
      float s = 0.f;
      for (int d = 0; d < HD; ++d)
        for (int r = 0; r < 8; ++r) {
          float t = w_rpe[(h * HD + d) * 40 + r * KD + kk];
          s += t * t;
        }
      w2[idx] = s * (1.0f / HD);
    }
    return;
  }
  int i = blockIdx.x * 256 + threadIdx.x;
  const int W = HDIM * 64;  // 32768
  if (i < W)                 owq[i] = (_Float16)wq[i];
  else if (i < 2 * W)        owk[i - W] = (_Float16)wk[i - W];
  else if (i < 3 * W)        owv[i - 2 * W] = (_Float16)wv[i - 2 * W];
  else if (i < 4 * W)        owo[i - 3 * W] = (_Float16)wo[i - 3 * W];
  else if (i < 4 * W + 4096) of1[i - 4 * W] = (_Float16)f1[i - 4 * W];
  else if (i < 4 * W + 8192) of2[i - 4 * W - 4096] = (_Float16)f2[i - 4 * W - 4096];
}

// LayerNorm over rows of 64, wave-per-row (wave32: 2 elems/lane), f16 output.
__global__ __launch_bounds__(256) void ln_cast_kernel(
    const float* __restrict__ x, const float* __restrict__ g,
    const float* __restrict__ b, _Float16* __restrict__ out, int nrows) {
  int lane = threadIdx.x & 31;
  int row = blockIdx.x * 8 + (threadIdx.x >> 5);
  if (row >= nrows) return;
  const float* xr = x + (size_t)row * 64;
  float v0 = xr[lane], v1 = xr[lane + 32];
  float s = v0 + v1;
#pragma unroll
  for (int off = 16; off > 0; off >>= 1) s += __shfl_xor(s, off, 32);
  float mean = s * (1.f / 64.f);
  float d0 = v0 - mean, d1 = v1 - mean;
  float vs = d0 * d0 + d1 * d1;
#pragma unroll
  for (int off = 16; off > 0; off >>= 1) vs += __shfl_xor(vs, off, 32);
  float rstd = rsqrtf(vs * (1.f / 64.f) + 1e-5f);
  out[(size_t)row * 64 + lane]      = (_Float16)(d0 * rstd * g[lane] + b[lane]);
  out[(size_t)row * 64 + lane + 32] = (_Float16)(d1 * rstd * g[lane + 32] + b[lane + 32]);
}

// ---------------------------------------------------------------------------
// LSH sort: hash + full bitonic sort of 2048-key chunks in LDS (phase 1).
// ---------------------------------------------------------------------------
__global__ __launch_bounds__(1024) void bitonic_local_sort_kernel(
    const float* __restrict__ coords, const float* __restrict__ hashp,
    unsigned long long* __restrict__ keys) {
  __shared__ unsigned long long lk[SORT_CHUNK];
  int base = blockIdx.x * SORT_CHUNK;
#pragma unroll
  for (int e = 0; e < 2; ++e) {
    int i = base + threadIdx.x + e * 1024;
    float s = 0.f;
#pragma unroll
    for (int k = 0; k < 6; ++k) s += coords[(size_t)i * 6 + k] * hashp[k];
    unsigned u = __float_as_uint(s);
    u = (u & 0x80000000u) ? ~u : (u | 0x80000000u);  // monotone map
    lk[threadIdx.x + e * 1024] = ((unsigned long long)u << 32) | (unsigned)i;
  }
  __syncthreads();
  for (int k = 2; k <= SORT_CHUNK; k <<= 1) {
    for (int j = k >> 1; j > 0; j >>= 1) {
      int i = threadIdx.x;
      int idx = ((i & ~(j - 1)) << 1) | (i & (j - 1));
      int ixj = idx | j;
      bool asc = (((base + idx) & k) == 0);
      unsigned long long a = lk[idx], b = lk[ixj];
      if ((a > b) == asc) { lk[idx] = b; lk[ixj] = a; }
      __syncthreads();
    }
  }
  keys[base + threadIdx.x] = lk[threadIdx.x];
  keys[base + threadIdx.x + 1024] = lk[threadIdx.x + 1024];
}

// Global compare-exchange step (strides j >= 2048), NPTS/2 threads.
__global__ void bitonic_step_kernel(unsigned long long* __restrict__ keys,
                                    int j, int k) {
  int t = blockIdx.x * 256 + threadIdx.x;
  int idx = ((t & ~(j - 1)) << 1) | (t & (j - 1));
  int ixj = idx | j;
  bool asc = ((idx & k) == 0);
  unsigned long long a = keys[idx], b = keys[ixj];
  if ((a > b) == asc) { keys[idx] = b; keys[ixj] = a; }
}

// All strides j = 1024..1 of merge stage k, within LDS chunks.
__global__ __launch_bounds__(1024) void bitonic_local_merge_kernel(
    unsigned long long* __restrict__ keys, int k) {
  __shared__ unsigned long long lk[SORT_CHUNK];
  int base = blockIdx.x * SORT_CHUNK;
  lk[threadIdx.x] = keys[base + threadIdx.x];
  lk[threadIdx.x + 1024] = keys[base + threadIdx.x + 1024];
  __syncthreads();
  bool asc = ((base & k) == 0);  // uniform per chunk since k >= 4096
  for (int j = 1024; j > 0; j >>= 1) {
    int i = threadIdx.x;
    int idx = ((i & ~(j - 1)) << 1) | (i & (j - 1));
    int ixj = idx | j;
    unsigned long long a = lk[idx], b = lk[ixj];
    if ((a > b) == asc) { lk[idx] = b; lk[ixj] = a; }
    __syncthreads();
  }
  keys[base + threadIdx.x] = lk[threadIdx.x];
  keys[base + threadIdx.x + 1024] = lk[threadIdx.x + 1024];
}

// ---------------------------------------------------------------------------
// Fused QKV + block attention: grid (512 blocks, 8 heads), 256 threads = 8 waves.
// Gathers xn rows (L2-resident, async global->LDS), computes q/k/v head slices
// in-kernel with WMMA, then scores (+RPE bias), softmax, AV, coalesced scatter.
// ---------------------------------------------------------------------------
__global__ __launch_bounds__(256) void attn_kernel(
    const _Float16* __restrict__ xn, const _Float16* __restrict__ wqw,
    const _Float16* __restrict__ wkw, const _Float16* __restrict__ wvw,
    const unsigned long long* __restrict__ keys,
    const float* __restrict__ coords, const float* __restrict__ w2,
    _Float16* __restrict__ ao /* [N, 512] in ORIGINAL row order */) {
  __shared__ _Float16 xl[BLK * 64];     // gathered xn tile
  __shared__ _Float16 qs[BLK * 64];     // q tile; reused to stage output
  __shared__ _Float16 ksm[BLK * 64];
  __shared__ _Float16 vT[64 * BLK];
  __shared__ _Float16 smat[BLK * BLK];  // scores -> attn (in place)
  __shared__ float csh[BLK * KD];
  __shared__ int ords[BLK];

  int blk = blockIdx.x;
  int h = blockIdx.y;
  int tid = threadIdx.x;

  if (tid < BLK) {
    int o = (int)(keys[(size_t)blk * BLK + tid] & 0xffffffffULL);
    ords[tid] = o;
#pragma unroll
    for (int kk = 0; kk < KD; ++kk)
      csh[tid * KD + kk] = coords[(size_t)o * 6 + kk];
  }
  __syncthreads();

  int row = tid >> 1, half = tid & 1;
  {  // async gather of xn rows: 64B per thread (4x b128 global->LDS)
    int o = ords[row];
    unsigned long long gx =
        (unsigned long long)(uintptr_t)(xn + (size_t)o * 64 + half * 32);
    unsigned lx = (unsigned)(uintptr_t)(xl + row * 64 + half * 32);
    asm volatile(
        "global_load_async_to_lds_b128 %0, %1, off\n\t"
        "global_load_async_to_lds_b128 %0, %1, off offset:16\n\t"
        "global_load_async_to_lds_b128 %0, %1, off offset:32\n\t"
        "global_load_async_to_lds_b128 %0, %1, off offset:48\n\t"
        :
        : "v"(lx), "v"(gx)
        : "memory");
    asm volatile("s_wait_asynccnt 0" ::: "memory");
  }
  __syncthreads();

  int wave = tid >> 5;
  int l = tid & 31;

  // q/k/v head slices: wave owns row tile `wave`; 4 col tiles; K=64.
  {
    const _Float16* wqs = wqw + (size_t)h * 64 * 64;
    const _Float16* wks = wkw + (size_t)h * 64 * 64;
    const _Float16* wvs = wvw + (size_t)h * 64 * 64;
    v16h a0 = frag_a(xl + wave * 16 * 64, 64);
    v16h a1 = frag_a(xl + wave * 16 * 64 + 32, 64);
#pragma unroll
    for (int tn = 0; tn < 4; ++tn) {
      int n0 = tn * 16;
      v8f cq = {}, ck = {}, cv = {};
      cq = WMMA_F16(a0, frag_b(wqs + (size_t)n0 * 64, 64), cq);
      cq = WMMA_F16(a1, frag_b(wqs + (size_t)n0 * 64 + 32, 64), cq);
      ck = WMMA_F16(a0, frag_b(wks + (size_t)n0 * 64, 64), ck);
      ck = WMMA_F16(a1, frag_b(wks + (size_t)n0 * 64 + 32, 64), ck);
      cv = WMMA_F16(a0, frag_b(wvs + (size_t)n0 * 64, 64), cv);
      cv = WMMA_F16(a1, frag_b(wvs + (size_t)n0 * 64 + 32, 64), cv);
#pragma unroll
      for (int r = 0; r < 8; ++r) {
        int mm = wave * 16 + r + 8 * (l >> 4);
        int nn = n0 + (l & 15);
        qs[mm * 64 + nn] = (_Float16)cq[r];
        ksm[mm * 64 + nn] = (_Float16)ck[r];
        vT[nn * BLK + mm] = (_Float16)cv[r];  // store v transposed
      }
    }
  }
  __syncthreads();

  float w2h[KD];
#pragma unroll
  for (int kk = 0; kk < KD; ++kk) w2h[kk] = w2[h * KD + kk];

  // scores: wave owns row tile; 8 col tiles; K=64 -> 2 WMMAs each.
  for (int tn = 0; tn < 8; ++tn) {
    v8f c = {};
#pragma unroll
    for (int k0 = 0; k0 < 64; k0 += 32) {
      v16h af = frag_a(qs + wave * 16 * 64 + k0, 64);
      v16h bf = frag_b(ksm + tn * 16 * 64 + k0, 64);
      c = WMMA_F16(af, bf, c);
    }
    int nn = tn * 16 + (l & 15);
#pragma unroll
    for (int r = 0; r < 8; ++r) {
      int mm = wave * 16 + r + 8 * (l >> 4);
      float bias = 0.f;
#pragma unroll
      for (int kk = 0; kk < KD; ++kk) {
        float dlt = csh[mm * KD + kk] - csh[nn * KD + kk];
        bias += w2h[kk] * dlt * dlt;
      }
      smat[mm * BLK + nn] = (_Float16)(c[r] * 0.125f - bias);
    }
  }
  __syncthreads();

  // softmax: one thread per row, in place (f16 storage, f32 math).
  if (tid < BLK) {
    _Float16* rowp = smat + tid * BLK;
    float mx = -1e30f;
    for (int c2 = 0; c2 < BLK; ++c2) mx = fmaxf(mx, (float)rowp[c2]);
    float s = 0.f;
    for (int c2 = 0; c2 < BLK; ++c2) s += __expf((float)rowp[c2] - mx);
    float inv = 1.f / s;
    for (int c2 = 0; c2 < BLK; ++c2)
      rowp[c2] = (_Float16)(__expf((float)rowp[c2] - mx) * inv);
  }
  __syncthreads();

  // out[128x64] = attn @ v; stage into qs (free now), then coalesced scatter.
  for (int tn = 0; tn < 4; ++tn) {
    v8f c = {};
#pragma unroll
    for (int k0 = 0; k0 < 128; k0 += 32) {
      v16h af = frag_a(smat + wave * 16 * BLK + k0, BLK);
      v16h bf = frag_b(vT + tn * 16 * BLK + k0, BLK);
      c = WMMA_F16(af, bf, c);
    }
#pragma unroll
    for (int r = 0; r < 8; ++r) {
      int mm = wave * 16 + r + 8 * (l >> 4);
      qs[mm * 64 + tn * 16 + (l & 15)] = (_Float16)c[r];
    }
  }
  __syncthreads();
  {
    int o = ords[row];
    uint4* dst = (uint4*)(ao + (size_t)o * HDIM + h * HD + half * 32);
    const uint4* src = (const uint4*)(qs + row * 64 + half * 32);
#pragma unroll
    for (int i = 0; i < 4; ++i) dst[i] = src[i];
  }
}

// ---------------------------------------------------------------------------
// Fused tail: y = x + ao @ w_out^T + b; h = LN2(y); out = y + relu(h@W1+b1)@W2+b2.
// All row-local: one 16-row tile per block, everything staged in LDS.
// ---------------------------------------------------------------------------
__global__ __launch_bounds__(128) void tail_kernel(
    const _Float16* __restrict__ ao, const _Float16* __restrict__ wout,
    const float* __restrict__ woutb, const float* __restrict__ x,
    const float* __restrict__ g2, const float* __restrict__ b2v,
    const _Float16* __restrict__ f1w, const float* __restrict__ f1b,
    const _Float16* __restrict__ f2w, const float* __restrict__ f2b,
    float* __restrict__ out) {
  __shared__ float sy[16 * 64];       // y tile (residual carried in LDS)
  __shared__ float so[16 * 64];       // final out tile
  __shared__ _Float16 sh[16 * 64];    // LN2(y) f16
  __shared__ _Float16 sf[16 * 64];    // relu(ffn1) f16
  int rt = blockIdx.x;
  int ct = threadIdx.x >> 5, l = threadIdx.x & 31;
  int n0 = ct * 16;

  // 1) w_out projection (K=512 -> 16 WMMAs per wave)
  {
    const _Float16* arow = ao + (size_t)rt * 16 * HDIM;
    v8f c = {};
#pragma unroll
    for (int k0 = 0; k0 < HDIM; k0 += 32) {
      v16h af = frag_a(arow + k0, HDIM);
      v16h bf = frag_b(wout + (size_t)n0 * HDIM + k0, HDIM);
      c = WMMA_F16(af, bf, c);
    }
    float bb = woutb[n0 + (l & 15)];
#pragma unroll
    for (int r = 0; r < 8; ++r)
      sy[(r + 8 * (l >> 4)) * 64 + n0 + (l & 15)] = c[r] + bb;
  }
  __syncthreads();

  // 2) add residual x (coalesced), keep y in LDS
  {
    int row = threadIdx.x >> 3, seg = threadIdx.x & 7;
    size_t gbase = ((size_t)rt * 16 + row) * 64 + seg * 8;
    const float4* xs = (const float4*)(x + gbase);
    float4* yv = (float4*)(sy + row * 64 + seg * 8);
#pragma unroll
    for (int i = 0; i < 2; ++i) {
      float4 a = yv[i], b = xs[i];
      yv[i] = make_float4(a.x + b.x, a.y + b.y, a.z + b.z, a.w + b.w);
    }
  }
  __syncthreads();

  // 3) LN2 per row: wave per row, 4 rows per iteration
  for (int rr = ct; rr < 16; rr += 4) {
    float v0 = sy[rr * 64 + l], v1 = sy[rr * 64 + l + 32];
    float s = v0 + v1;
#pragma unroll
    for (int off = 16; off > 0; off >>= 1) s += __shfl_xor(s, off, 32);
    float mean = s * (1.f / 64.f);
    float d0 = v0 - mean, d1 = v1 - mean;
    float vs = d0 * d0 + d1 * d1;
#pragma unroll
    for (int off = 16; off > 0; off >>= 1) vs += __shfl_xor(vs, off, 32);
    float rstd = rsqrtf(vs * (1.f / 64.f) + 1e-5f);
    sh[rr * 64 + l]      = (_Float16)(d0 * rstd * g2[l] + b2v[l]);
    sh[rr * 64 + l + 32] = (_Float16)(d1 * rstd * g2[l + 32] + b2v[l + 32]);
  }
  __syncthreads();

  // 4) FFN1 (relu fused)
  {
    v8f c = {};
#pragma unroll
    for (int k0 = 0; k0 < 64; k0 += 32) {
      v16h af = frag_a(sh + k0, 64);
      v16h bf = frag_b(f1w + (size_t)n0 * 64 + k0, 64);
      c = WMMA_F16(af, bf, c);
    }
    float bb = f1b[n0 + (l & 15)];
#pragma unroll
    for (int r = 0; r < 8; ++r)
      sf[(r + 8 * (l >> 4)) * 64 + n0 + (l & 15)] =
          (_Float16)fmaxf(c[r] + bb, 0.f);
  }
  __syncthreads();

  // 5) FFN2 + residual y
  {
    v8f c = {};
#pragma unroll
    for (int k0 = 0; k0 < 64; k0 += 32) {
      v16h af = frag_a(sf + k0, 64);
      v16h bf = frag_b(f2w + (size_t)n0 * 64 + k0, 64);
      c = WMMA_F16(af, bf, c);
    }
    float bb = f2b[n0 + (l & 15)];
#pragma unroll
    for (int r = 0; r < 8; ++r) {
      int mm = r + 8 * (l >> 4);
      int nn = n0 + (l & 15);
      so[mm * 64 + nn] = c[r] + bb + sy[mm * 64 + nn];
    }
  }
  __syncthreads();

  // 6) coalesced store
  {
    int row = threadIdx.x >> 3, seg = threadIdx.x & 7;
    size_t gbase = ((size_t)rt * 16 + row) * 64 + seg * 8;
    const float4* src = (const float4*)(so + row * 64 + seg * 8);
    float4* dst = (float4*)(out + gbase);
#pragma unroll
    for (int i = 0; i < 2; ++i) dst[i] = src[i];
  }
}

// ---------------------------------------------------------------------------
extern "C" void kernel_launch(void* const* d_in, const int* in_sizes, int n_in,
                              void* d_out, int out_size, void* d_ws, size_t ws_size,
                              hipStream_t stream) {
  (void)in_sizes; (void)n_in; (void)out_size; (void)ws_size;
  const float* x        = (const float*)d_in[0];
  const float* coords   = (const float*)d_in[1];
  const float* wq       = (const float*)d_in[2];
  const float* wk       = (const float*)d_in[3];
  const float* wv       = (const float*)d_in[4];
  const float* w_rpe    = (const float*)d_in[5];
  const float* w_out_w  = (const float*)d_in[6];
  const float* w_out_b  = (const float*)d_in[7];
  const float* ln1_g    = (const float*)d_in[8];
  const float* ln1_b    = (const float*)d_in[9];
  const float* ln2_g    = (const float*)d_in[10];
  const float* ln2_b    = (const float*)d_in[11];
  const float* ff1_w    = (const float*)d_in[12];
  const float* ff1_b    = (const float*)d_in[13];
  const float* ff2_w    = (const float*)d_in[14];
  const float* ff2_b    = (const float*)d_in[15];
  const float* hashp    = (const float*)d_in[16];
  float* out = (float*)d_out;

  // workspace layout (256B aligned slots)
  size_t off = 0;
  auto alloc = [&](size_t bytes) {
    void* p = (char*)d_ws + off;
    off += (bytes + 255) & ~(size_t)255;
    return p;
  };
  _Float16* xn_h  = (_Float16*)alloc((size_t)NPTS * 64 * 2);
  _Float16* ao_h  = (_Float16*)alloc((size_t)NPTS * HDIM * 2);
  unsigned long long* keys = (unsigned long long*)alloc((size_t)NPTS * 8);
  _Float16* wq_h   = (_Float16*)alloc((size_t)HDIM * 64 * 2);
  _Float16* wk_h   = (_Float16*)alloc((size_t)HDIM * 64 * 2);
  _Float16* wv_h   = (_Float16*)alloc((size_t)HDIM * 64 * 2);
  _Float16* wout_h = (_Float16*)alloc((size_t)64 * HDIM * 2);
  _Float16* ff1_h  = (_Float16*)alloc((size_t)64 * 64 * 2);
  _Float16* ff2_h  = (_Float16*)alloc((size_t)64 * 64 * 2);
  float* w2 = (float*)alloc(HEADS * KD * 4);

  // 1) weights -> f16 + w2 reduction (single launch; last block does w2)
  int conv_n = 4 * HDIM * 64 + 2 * 64 * 64;
  int conv_blocks = (conv_n + 255) / 256 + 1;
  prep_kernel<<<conv_blocks, 256, 0, stream>>>(
      wq, wk, wv, w_out_w, ff1_w, ff2_w, w_rpe,
      wq_h, wk_h, wv_h, wout_h, ff1_h, ff2_h, w2);

  // 2) LN1 -> xn (f16)
  ln_cast_kernel<<<NPTS / 8, 256, 0, stream>>>(x, ln1_g, ln1_b, xn_h, NPTS);

  // 3) LSH sort: hash fused into LDS-chunked bitonic phase 1
  bitonic_local_sort_kernel<<<NPTS / SORT_CHUNK, 1024, 0, stream>>>(
      coords, hashp, keys);
  for (int k = 2 * SORT_CHUNK; k <= NPTS; k <<= 1) {
    for (int j = k >> 1; j >= SORT_CHUNK; j >>= 1)
      bitonic_step_kernel<<<NPTS / 512, 256, 0, stream>>>(keys, j, k);
    bitonic_local_merge_kernel<<<NPTS / SORT_CHUNK, 1024, 0, stream>>>(keys, k);
  }

  // 4) fused QKV + block attention (WMMA + async gather of L2-resident xn)
  attn_kernel<<<dim3(NPTS / BLK, HEADS), 256, 0, stream>>>(
      xn_h, wq_h, wk_h, wv_h, keys, coords, w2, ao_h);

  // 5) fused tail: w_out proj + residual + LN2 + FFN + residual
  tail_kernel<<<NPTS / 16, 128, 0, stream>>>(
      ao_h, wout_h, w_out_b, x, ln2_g, ln2_b, ff1_h, ff1_b, ff2_h, ff2_b, out);
}
